// RNNModel_4810363372934
// MI455X (gfx1250) — compile-verified, matrix operand order checked by reading, not run on hardware
//
#include <hip/hip_runtime.h>

typedef __attribute__((ext_vector_type(16))) _Float16 v16h;
typedef __attribute__((ext_vector_type(8)))  float    v8f;

#define HDIM 10

// lane <-> lane^16 exchange. Prefer v_permlanex16_b32 (VALU, no DScnt wait)
// over ds_swizzle (LDS pipe, forces s_wait_dscnt before the next WMMA).
__device__ __forceinline__ float xhalf(float v) {
#if __has_builtin(__builtin_amdgcn_permlanex16)
    const int i = __builtin_bit_cast(int, v);
    const int r = __builtin_amdgcn_permlanex16(i, i, 0x76543210, (int)0xfedcba98,
                                               false, false);
    return __builtin_bit_cast(float, r);
#else
    return __builtin_bit_cast(float,
        __builtin_amdgcn_ds_swizzle(__builtin_bit_cast(int, v), 0x401F));
#endif
}

__device__ __forceinline__ float fast_tanh(float x) {
#if __has_builtin(__builtin_amdgcn_tanhf)
    return __builtin_amdgcn_tanhf(x);             // v_tanh_f32 (confirmed round 1)
#else
    float e = __builtin_amdgcn_exp2f(x * 2.8853900817779268f);
    return 1.0f - 2.0f * __builtin_amdgcn_rcpf(e + 1.0f);
#endif
}

// One wave advances TWO independent 16-chain tiles (ILP=2 to fill the
// WMMA-hazard / tanh latency shadow of the serial recurrence).
// Per tile per step: D = A(W_hh 16x32 f16) * B(h 32x16 f16) + C(x*w_in + bias).
// C/D layout: lanes 0-15 -> N=batch, VGPR r -> M = r + 8*(lane>=16).
__global__ void __launch_bounds__(32)
rnn_scan_wmma(const float* __restrict__ x,
              const float* __restrict__ W_ih,
              const float* __restrict__ W_hh,
              const float* __restrict__ b_ih,
              const float* __restrict__ b_hh,
              const float* __restrict__ fc_W,
              const float* __restrict__ fc_b,
              float* __restrict__ out,
              int T, int Bsz)
{
    const int lane = threadIdx.x & 31;
    const int hi   = lane >> 4;
    const int bl   = lane & 15;
    const int b0   = blockIdx.x * 32 + bl;   // tile 0 batch row
    const int b1   = b0 + 16;                // tile 1 batch row
    const int br0  = (b0 < Bsz) ? b0 : (Bsz - 1);
    const int br1  = (b1 < Bsz) ? b1 : (Bsz - 1);

    // ---- constant A = W_hh, zero-padded to 16x32, f16 A-layout ----
    // element e: VGPR v=e/2, slot s=e%2, K = 16*(v/4) + 2*(v%4) + s + 8*hi, M = bl
    v16h A;
#pragma unroll
    for (int e = 0; e < 16; ++e) {
        const int v = e >> 1, s = e & 1;
        const int K = 16 * (v >> 2) + 2 * (v & 3) + s + 8 * hi;
        const float w = (bl < HDIM && K < HDIM) ? W_hh[bl * HDIM + K] : 0.0f;
        A[e] = (_Float16)w;
    }

    // per-lane C-layout constants (row j = r + 8*hi)
    float wv[8], bv[8], fcv[8];
#pragma unroll
    for (int r = 0; r < 8; ++r) {
        const int j = r + 8 * hi;
        const bool ok = (j < HDIM);
        wv[r]  = ok ? W_ih[j] : 0.0f;
        bv[r]  = ok ? (b_ih[j] + b_hh[j]) : 0.0f;
        fcv[r] = ok ? fc_W[j] : 0.0f;
    }

    const float* xrow0 = x + (size_t)br0 * (size_t)T;
    const float* xrow1 = x + (size_t)br1 * (size_t)T;

    v16h Bm0 = {}, Bm1 = {};           // h = 0
    float h0[8], h1[8];
#pragma unroll
    for (int r = 0; r < 8; ++r) { h0[r] = 0.0f; h1[r] = 0.0f; }

    auto step = [&](float xt0, float xt1) {
        v8f c0, c1;
#pragma unroll
        for (int r = 0; r < 8; ++r) {
            c0[r] = __builtin_fmaf(xt0, wv[r], bv[r]);
            c1[r] = __builtin_fmaf(xt1, wv[r], bv[r]);
        }
        c0 = __builtin_amdgcn_wmma_f32_16x16x32_f16(false, A, false, Bm0,
                                                    (short)0, c0, false, false);
        c1 = __builtin_amdgcn_wmma_f32_16x16x32_f16(false, A, false, Bm1,
                                                    (short)0, c1, false, false);
        // Pull rows j=8,9 (upper-half lanes of D VGPRs 0,1) BEFORE tanh so the
        // permute overlaps the tanh chain (swap/tanh commute pointwise).
        const float e80 = xhalf(c0[0]), e90 = xhalf(c0[1]);
        const float e81 = xhalf(c1[0]), e91 = xhalf(c1[1]);
#pragma unroll
        for (int r = 0; r < 8; ++r) {
            h0[r] = fast_tanh(c0[r]);
            h1[r] = fast_tanh(c1[r]);
        }
        const float h80 = fast_tanh(e80), h90 = fast_tanh(e90);
        const float h81 = fast_tanh(e81), h91 = fast_tanh(e91);
        // Rebuild B (elems 0-7 hold K=8*hi+r, K=0..9 real). Elems 8,9 carry junk but A cols K>=10 are
        // zero and tanh output is finite, so those products are exactly 0.
#pragma unroll
        for (int r = 0; r < 8; ++r) { Bm0[r] = (_Float16)h0[r]; Bm1[r] = (_Float16)h1[r]; }
        Bm0[8] = (_Float16)h80; Bm0[9] = (_Float16)h90;
        Bm1[8] = (_Float16)h81; Bm1[9] = (_Float16)h91;
        // Bm*[10..15] stay zero from init
    };

    int t = 0;
    if ((T & 3) == 0) {
        for (; t < T; t += 4) {
            const float4 xq0 = *(const float4*)(xrow0 + t);
            const float4 xq1 = *(const float4*)(xrow1 + t);
            __builtin_prefetch(xrow0 + t + 256, 0, 3);   // global_prefetch_b8, near
            __builtin_prefetch(xrow1 + t + 256, 0, 3);
            step(xq0.x, xq1.x); step(xq0.y, xq1.y);
            step(xq0.z, xq1.z); step(xq0.w, xq1.w);
        }
    }
    for (; t < T; ++t) step(xrow0[t], xrow1[t]);

    // out[b] = fc_W . h_last + fc_b ; halves j=0..7 / j=8..9 summed via xhalf
    float p0 = 0.0f, p1 = 0.0f;
#pragma unroll
    for (int r = 0; r < 8; ++r) {
        p0 = __builtin_fmaf(h0[r], fcv[r], p0);
        p1 = __builtin_fmaf(h1[r], fcv[r], p1);
    }
    p0 += xhalf(p0);
    p1 += xhalf(p1);
    if (hi == 0) {
        const float fb = fc_b[0];
        if (b0 < Bsz) out[b0] = p0 + fb;
        if (b1 < Bsz) out[b1] = p1 + fb;
    }
}

extern "C" void kernel_launch(void* const* d_in, const int* in_sizes, int n_in,
                              void* d_out, int out_size, void* d_ws, size_t ws_size,
                              hipStream_t stream) {
    const float* x   = (const float*)d_in[0];
    const float* Wih = (const float*)d_in[1];
    const float* Whh = (const float*)d_in[2];
    const float* bih = (const float*)d_in[3];
    const float* bhh = (const float*)d_in[4];
    const float* fcW = (const float*)d_in[5];
    const float* fcb = (const float*)d_in[6];
    float* out = (float*)d_out;

    const int Bsz = out_size;               // 8192
    const int T   = in_sizes[0] / Bsz;      // 4096
    const int blocks = (Bsz + 31) / 32;     // 256 waves, 2x16 chains each

    rnn_scan_wmma<<<blocks, 32, 0, stream>>>(x, Wih, Whh, bih, bhh, fcW, fcb,
                                             out, T, Bsz);
}
